// MultiHeadAttention_60198261621223
// MI455X (gfx1250) — matrix operand order repaired and checked by reading, not compiled, not run on hardware
//
#include <hip/hip_runtime.h>
#include <hip/hip_bf16.h>

// MHA forward for B=1, S=4096, D=1024, H=16, HD=64 on gfx1250 (wave32, WMMA).
// Pipeline: cast/transpose -> QKV proj (register-blocked WMMA f16->f32) ->
// causal flash attention (S^T trick, no score materialization) ->
// output projection (register-blocked WMMA, fp32 out).
// Workspace layout (48 MiB total): xh | WqT WkT WvT WoT | Q | K | Vt | O

#define S_LEN 4096
#define D_DIM 1024
#define NH    16
#define HD    64

typedef _Float16 v8h  __attribute__((ext_vector_type(8)));
typedef _Float16 v16h __attribute__((ext_vector_type(16)));
typedef float    v8f  __attribute__((ext_vector_type(8)));

__device__ __forceinline__ v16h cat8(v8h lo, v8h hi) {
    return __builtin_shufflevector(lo, hi, 0,1,2,3,4,5,6,7,8,9,10,11,12,13,14,15);
}

// A fragment (16x32 f16, row-major source). `A` pre-offset to (tile row 0, k0).
// Lane L: m=L&15, half=L>>4. v16h element e holds K = (e<8 ? e : e+8) + half*8.
__device__ __forceinline__ v16h load_a(const _Float16* __restrict__ A, int ld, int lane) {
    int m = lane & 15, hf = lane >> 4;
    const _Float16* p = A + (size_t)m * ld;
    v8h lo = *(const v8h*)(p + hf * 8);
    v8h hi = *(const v8h*)(p + 16 + hf * 8);
    return cat8(lo, hi);
}

// B fragment (32x16 f16) from B^T row-major [N x K]. `BT` pre-offset to (n0, k0).
// Lane L: n=L&15, half=L>>4. v16h element e holds K = (e<8 ? e : e+8) + half*8.
__device__ __forceinline__ v16h load_bt(const _Float16* __restrict__ BT, int ld, int lane) {
    int n = lane & 15, hf = lane >> 4;
    const _Float16* p = BT + (size_t)n * ld;
    v8h lo = *(const v8h*)(p + hf * 8);
    v8h hi = *(const v8h*)(p + 16 + hf * 8);
    return cat8(lo, hi);
}

__device__ __forceinline__ v8f wmma32(v16h a, v16h b, v8f c) {
    return __builtin_amdgcn_wmma_f32_16x16x32_f16(false, a, false, b, (short)0, c, false, false);
}

// ---------------- conversion kernels ----------------
__global__ void cvt_f32_f16(const float* __restrict__ src, _Float16* __restrict__ dst, int n) {
    int i = blockIdx.x * 256 + threadIdx.x;
    if (i < n) dst[i] = (_Float16)src[i];
}

// WT[n][k] = (f16) W[k][n], 1024x1024
__global__ void transpose_w_f16(const float* __restrict__ W, _Float16* __restrict__ WT) {
    int i = blockIdx.x * 256 + threadIdx.x;
    int n = i >> 10, k = i & 1023;
    WT[i] = (_Float16)W[(size_t)k * D_DIM + n];
}

// ---------------- QKV projection (register-blocked 32x64 per wave) ----------------
// out = xh @ W + b ; vmode==0 -> [h][s][hd] (Q,K), vmode==1 -> [h][hd][s] (V^T)
// Per K-step: 2 A frags + 4 B frags feed 8 WMMAs (1.5 b128 loads per WMMA).
__global__ void proj_qkv(const _Float16* __restrict__ xh,
                         const _Float16* __restrict__ WT,
                         const float* __restrict__ bias,
                         _Float16* __restrict__ out, int vmode) {
    int lane = threadIdx.x & 31;
    int wave = threadIdx.x >> 5;
    int tile = blockIdx.x * 8 + wave;       // 128 (M/32) x 16 (N/64) wave tiles
    int m0 = (tile >> 4) * 32;
    int n0 = (tile & 15) * 64;
    const _Float16* a0base = xh + (size_t)m0 * D_DIM;
    const _Float16* a1base = xh + (size_t)(m0 + 16) * D_DIM;
    const _Float16* bbase  = WT + (size_t)n0 * D_DIM;
    v8f acc[2][4] = {};
    for (int k0 = 0; k0 < D_DIM; k0 += 32) {
        __builtin_prefetch(a0base + k0 + 64, 0, 3);   // WGP-scope prefetch
        v16h a0 = load_a(a0base + k0, D_DIM, lane);
        v16h a1 = load_a(a1base + k0, D_DIM, lane);
#pragma unroll
        for (int j = 0; j < 4; ++j) {
            v16h b = load_bt(bbase + (size_t)(j * 16) * D_DIM + k0, D_DIM, lane);
            acc[0][j] = wmma32(a0, b, acc[0][j]);
            acc[1][j] = wmma32(a1, b, acc[1][j]);
        }
    }
    int n = lane & 15, hf = lane >> 4;
#pragma unroll
    for (int j = 0; j < 4; ++j) {
        int col = n0 + j * 16 + n;
        float bv = bias[col];
        int h = col >> 6, hd = col & 63;
#pragma unroll
        for (int i = 0; i < 2; ++i) {
#pragma unroll
            for (int r = 0; r < 8; ++r) {
                int m = m0 + i * 16 + r + 8 * hf;
                float v = acc[i][j][r] + bv;
                if (vmode)
                    out[(size_t)h * HD * S_LEN + (size_t)hd * S_LEN + m] = (_Float16)v;
                else
                    out[(size_t)h * S_LEN * HD + (size_t)m * HD + hd] = (_Float16)v;
            }
        }
    }
}

// ---------------- causal flash attention ----------------
// One wave per (head, 16-row query tile). Scores computed transposed:
// T = K_tile @ Q^T so C-layout lane index == query row; probabilities re-pack
// lane-locally into the A fragment of the P@V WMMA.
__global__ void attn_fwd(const _Float16* __restrict__ Q,
                         const _Float16* __restrict__ K,
                         const _Float16* __restrict__ Vt,
                         _Float16* __restrict__ O) {
    int lane = threadIdx.x & 31;
    int wave = threadIdx.x >> 5;
    int gw = blockIdx.x * 8 + wave;   // 4096 waves total
    int h  = gw >> 8;                 // 256 q-tiles per head
    int q0 = (gw & 255) * 16;
    int n = lane & 15, hf = lane >> 4;

    const _Float16* Qh = Q  + (size_t)h * S_LEN * HD;
    const _Float16* Kh = K  + (size_t)h * S_LEN * HD;
    const _Float16* Vh = Vt + (size_t)h * HD * S_LEN;

    // B fragments for S^T: B = Q^T (B^T layout == Q row-major [q][hd])
    v16h bq0 = load_bt(Qh + (size_t)q0 * HD, HD, lane);
    v16h bq1 = load_bt(Qh + (size_t)q0 * HD + 32, HD, lane);

    v8f o0 = {}, o1 = {}, o2 = {}, o3 = {};
    float mrun = -3.0e38f, lrun = 0.0f;
    const float sc = 0.125f * 1.44269504089f;  // 1/sqrt(64) folded with log2(e)

    for (int j0 = 0; j0 < q0 + 16; j0 += 32) {
        // S^T tiles: T0 = seq[j0..j0+15] x q, T1 = seq[j0+16..j0+31] x q, K-dim = 64
        v8f t0 = {}, t1 = {};
        {
            v16h a = load_a(Kh + (size_t)j0 * HD, HD, lane);
            t0 = wmma32(a, bq0, t0);
            a = load_a(Kh + (size_t)j0 * HD + 32, HD, lane);
            t0 = wmma32(a, bq1, t0);
            a = load_a(Kh + (size_t)(j0 + 16) * HD, HD, lane);
            t1 = wmma32(a, bq0, t1);
            a = load_a(Kh + (size_t)(j0 + 16) * HD + 32, HD, lane);
            t1 = wmma32(a, bq1, t1);
        }
        int q = q0 + n;
        bool dm = (j0 + 31 > q0);  // only diagonal-touching blocks need masking
        float s0[8], s1[8];
#pragma unroll
        for (int r = 0; r < 8; ++r) {
            int seq0 = j0 + r + 8 * hf;
            float x0 = t0[r] * sc;
            float x1 = t1[r] * sc;
            if (dm) {
                if (seq0 > q)      x0 = -3.0e38f;
                if (seq0 + 16 > q) x1 = -3.0e38f;
            }
            s0[r] = x0; s1[r] = x1;
        }
        // online softmax stats (base-2): in-lane max/sum + xor-16 exchange
        float mloc = s0[0];
#pragma unroll
        for (int r = 1; r < 8; ++r) mloc = fmaxf(mloc, s0[r]);
#pragma unroll
        for (int r = 0; r < 8; ++r) mloc = fmaxf(mloc, s1[r]);
        mloc = fmaxf(mloc, __shfl_xor(mloc, 16, 32));
        float mnew  = fmaxf(mrun, mloc);
        float alpha = exp2f(mrun - mnew);
        float p0[8], p1[8], ls = 0.0f;
#pragma unroll
        for (int r = 0; r < 8; ++r) {
            p0[r] = exp2f(s0[r] - mnew);
            p1[r] = exp2f(s1[r] - mnew);
            ls += p0[r] + p1[r];
        }
        ls += __shfl_xor(ls, 16, 32);
        lrun = lrun * alpha + ls;
        mrun = mnew;
        // lane-local re-pack: P (16q x 32seq) A fragment
        v16h pa;
#pragma unroll
        for (int e = 0; e < 8; ++e) {
            pa[e]     = (_Float16)p0[e];
            pa[8 + e] = (_Float16)p1[e];
        }
        // rescale O accumulator rows (row m = r + 8*hf holds stats in lane r+8*hf)
        float ar[8];
#pragma unroll
        for (int r = 0; r < 8; ++r) ar[r] = __shfl(alpha, r + 8 * hf, 32);
#pragma unroll
        for (int r = 0; r < 8; ++r) {
            o0[r] *= ar[r]; o1[r] *= ar[r]; o2[r] *= ar[r]; o3[r] *= ar[r];
        }
        // P @ V : B fragments from V^T [h][hd][s] (contiguous along seq)
        v16h b0 = load_bt(Vh + (size_t)(0 * 16) * S_LEN + j0, S_LEN, lane);
        v16h b1 = load_bt(Vh + (size_t)(1 * 16) * S_LEN + j0, S_LEN, lane);
        v16h b2 = load_bt(Vh + (size_t)(2 * 16) * S_LEN + j0, S_LEN, lane);
        v16h b3 = load_bt(Vh + (size_t)(3 * 16) * S_LEN + j0, S_LEN, lane);
        o0 = wmma32(pa, b0, o0);
        o1 = wmma32(pa, b1, o1);
        o2 = wmma32(pa, b2, o2);
        o3 = wmma32(pa, b3, o3);
    }
    float rinv = 1.0f / lrun;
    float lr[8];
#pragma unroll
    for (int r = 0; r < 8; ++r) lr[r] = __shfl(rinv, r + 8 * hf, 32);
    _Float16* Oh = O + (size_t)h * S_LEN * HD;
#pragma unroll
    for (int r = 0; r < 8; ++r) {
        int m = q0 + r + 8 * hf;
        Oh[(size_t)m * HD + 0 * 16 + n] = (_Float16)(o0[r] * lr[r]);
        Oh[(size_t)m * HD + 1 * 16 + n] = (_Float16)(o1[r] * lr[r]);
        Oh[(size_t)m * HD + 2 * 16 + n] = (_Float16)(o2[r] * lr[r]);
        Oh[(size_t)m * HD + 3 * 16 + n] = (_Float16)(o3[r] * lr[r]);
    }
}

// ---------------- output projection (register-blocked 32x64 per wave) ----------------
// out[s][d] = concat_heads(O) @ Wo + bo, fp32. O is head-major [h][s][hd];
// since k-steps (32) never straddle a 64-wide head block, A loads stay contiguous.
__global__ void out_proj(const _Float16* __restrict__ O,
                         const _Float16* __restrict__ WoT,
                         const float* __restrict__ bo,
                         float* __restrict__ out) {
    int lane = threadIdx.x & 31;
    int wave = threadIdx.x >> 5;
    int tile = blockIdx.x * 8 + wave;       // 128 x 16 wave tiles
    int m0 = (tile >> 4) * 32;
    int n0 = (tile & 15) * 64;
    int n = lane & 15, hf = lane >> 4;
    int ma0 = m0 + n;        // lane's A rows for the two M sub-tiles
    int ma1 = m0 + 16 + n;
    v8f acc[2][4] = {};
    for (int k0 = 0; k0 < D_DIM; k0 += 32) {
        int h = k0 >> 6, kh = k0 & 63;
        const _Float16* p0 = O + (size_t)h * S_LEN * HD + (size_t)ma0 * HD + kh;
        const _Float16* p1 = O + (size_t)h * S_LEN * HD + (size_t)ma1 * HD + kh;
        v16h a0 = cat8(*(const v8h*)(p0 + hf * 8), *(const v8h*)(p0 + 16 + hf * 8));
        v16h a1 = cat8(*(const v8h*)(p1 + hf * 8), *(const v8h*)(p1 + 16 + hf * 8));
#pragma unroll
        for (int j = 0; j < 4; ++j) {
            v16h b = load_bt(WoT + (size_t)(n0 + j * 16) * D_DIM + k0, D_DIM, lane);
            acc[0][j] = wmma32(a0, b, acc[0][j]);
            acc[1][j] = wmma32(a1, b, acc[1][j]);
        }
    }
#pragma unroll
    for (int j = 0; j < 4; ++j) {
        int col = n0 + j * 16 + n;
        float bv = bo[col];
#pragma unroll
        for (int i = 0; i < 2; ++i) {
#pragma unroll
            for (int r = 0; r < 8; ++r) {
                int mm = m0 + i * 16 + r + 8 * hf;
                out[(size_t)mm * D_DIM + col] = acc[i][j][r] + bv;
            }
        }
    }
}

// ---------------- host launch ----------------
extern "C" void kernel_launch(void* const* d_in, const int* in_sizes, int n_in,
                              void* d_out, int out_size, void* d_ws, size_t ws_size,
                              hipStream_t stream) {
    const float* x  = (const float*)d_in[0];
    const float* Wq = (const float*)d_in[1];
    const float* bq = (const float*)d_in[2];
    const float* Wk = (const float*)d_in[3];
    const float* bk = (const float*)d_in[4];
    const float* Wv = (const float*)d_in[5];
    const float* bv = (const float*)d_in[6];
    const float* Wo = (const float*)d_in[7];
    const float* bo = (const float*)d_in[8];
    float* out = (float*)d_out;

    char* ws = (char*)d_ws;
    const size_t MB = 1 << 20;
    _Float16* xh  = (_Float16*)(ws + 0 * MB);   // 8 MiB
    _Float16* WqT = (_Float16*)(ws + 8 * MB);   // 2 MiB each
    _Float16* WkT = (_Float16*)(ws + 10 * MB);
    _Float16* WvT = (_Float16*)(ws + 12 * MB);
    _Float16* WoT = (_Float16*)(ws + 14 * MB);
    _Float16* Qb  = (_Float16*)(ws + 16 * MB);  // 8 MiB, [h][s][hd]
    _Float16* Kb  = (_Float16*)(ws + 24 * MB);  // 8 MiB, [h][s][hd]
    _Float16* Vtb = (_Float16*)(ws + 32 * MB);  // 8 MiB, [h][hd][s]
    _Float16* Ob  = (_Float16*)(ws + 40 * MB);  // 8 MiB, [h][s][hd]

    // 1) casts / transposes
    cvt_f32_f16<<<(S_LEN * D_DIM) / 256, 256, 0, stream>>>(x, xh, S_LEN * D_DIM);
    transpose_w_f16<<<(D_DIM * D_DIM) / 256, 256, 0, stream>>>(Wq, WqT);
    transpose_w_f16<<<(D_DIM * D_DIM) / 256, 256, 0, stream>>>(Wk, WkT);
    transpose_w_f16<<<(D_DIM * D_DIM) / 256, 256, 0, stream>>>(Wv, WvT);
    transpose_w_f16<<<(D_DIM * D_DIM) / 256, 256, 0, stream>>>(Wo, WoT);

    // 2) QKV projections: 2048 32x64 wave tiles, 8 waves/block
    proj_qkv<<<256, 256, 0, stream>>>(xh, WqT, bq, Qb, 0);
    proj_qkv<<<256, 256, 0, stream>>>(xh, WkT, bk, Kb, 0);
    proj_qkv<<<256, 256, 0, stream>>>(xh, WvT, bv, Vtb, 1);

    // 3) causal flash attention: 16 heads * 256 q-tiles = 4096 waves
    attn_fwd<<<512, 256, 0, stream>>>(Qb, Kb, Vtb, Ob);

    // 4) output projection (fp32 result)
    out_proj<<<256, 256, 0, stream>>>(Ob, WoT, bo, out);
}
